// HybridGemmaDiT_76330158785255
// MI455X (gfx1250) — compile-verified, matrix operand order checked by reading, not run on hardware
//
#include <hip/hip_runtime.h>
#include <math.h>

// ---------------------------------------------------------------------------
// HybridGemmaDiT forward for MI455X (gfx1250, wave32, WMMA).
// All GEMMs run through v_wmma_f32_16x16x32_bf16 (bf16 operands, f32 accum):
// the model is compute-bound (~2.5 TFLOP vs ~1 GB HBM traffic per forward),
// so WMMA throughput is the ceiling, not bandwidth. K-tile prefetch via
// global_prefetch_b8 hides global latency behind the WMMA/barrier phase.
// ---------------------------------------------------------------------------

typedef __attribute__((ext_vector_type(16))) __bf16 bf16x16;
typedef __attribute__((ext_vector_type(8)))  float  f32x8;

union FragAB { bf16x16 v; unsigned u[8]; };

__device__ __forceinline__ unsigned short f2bf(float f) {
  unsigned u = __float_as_uint(f);
  unsigned r = u + 0x7FFFu + ((u >> 16) & 1u);   // round-to-nearest-even
  return (unsigned short)(r >> 16);
}

// CDNA5 16-bit A/B fragment K mapping (ISA 7.12.2):
// lanes 0-15: VGPR v<4 -> K=2v,2v+1 ; v>=4 -> K=16+2(v-4). lanes 16-31: +8.
__device__ __forceinline__ int kmap_(int v, int half) {
  return ((v < 4) ? (2 * v) : (16 + 2 * (v - 4))) + 8 * half;
}

__device__ __forceinline__ f32x8 zero8() {
  f32x8 z;
#pragma unroll
  for (int i = 0; i < 8; ++i) z[i] = 0.f;
  return z;
}

// ---------------------------------------------------------------------------
// Generic bf16 WMMA GEMM: C[M,N] = A[M,K] @ B[K,N]  (+epilogue)
// A,B,C f32 in global; converted to bf16 while staging into LDS.
// Block: 128 threads = 4 waves, each wave 32x32 (2x2 WMMA tiles). Tile 64x64.
// EPI: 0 store, 1 store+bias, 2 C += v+bias (residual), 3 C += aux*sigmoid(v+bias)
// ---------------------------------------------------------------------------
template <int EPI>
__global__ __launch_bounds__(128) void gemm_bf16_wmma(
    const float* __restrict__ A, long lda,
    const float* __restrict__ B, long ldb,
    const float* __restrict__ bias,
    float* __restrict__ C, long ldc,
    const float* __restrict__ aux, long ldaux,
    int M, int N, int K) {
  __shared__ __align__(8) unsigned short sA[64][36];   // [m][k]
  __shared__ __align__(8) unsigned short sB[64][36];   // [n][k] (transposed)
  const int tid  = threadIdx.x;
  const int lane = tid & 31, wid = tid >> 5;
  const int half = lane >> 4, lrow = lane & 15;
  const int wm = (wid >> 1) * 32, wn = (wid & 1) * 32;
  const long br = (long)blockIdx.y * 64, bc = (long)blockIdx.x * 64;

  f32x8 acc[2][2];
  acc[0][0] = zero8(); acc[0][1] = zero8();
  acc[1][0] = zero8(); acc[1][1] = zero8();

  const int ar = tid >> 1, ach = (tid & 1) * 16;       // A staging coords
  const int bk = tid >> 2, bnb = (tid & 3) * 16;       // B staging coords

  for (int k0 = 0; k0 < K; k0 += 32) {
    {  // stage A tile 64x32 (rows always in range: M % 64 == 0 in this net)
      const float* ap = A + (br + ar) * lda + (k0 + ach);
#pragma unroll
      for (int i = 0; i < 16; ++i) sA[ar][ach + i] = f2bf(ap[i]);
    }
    {  // stage B tile 32x64, transposed into [n][k]; zero-fill cols >= N
      const float* bp = B + (long)(k0 + bk) * ldb + bc + bnb;
#pragma unroll
      for (int i = 0; i < 16; ++i) {
        const long col = bc + bnb + i;
        sB[bnb + i][bk] = f2bf(col < N ? bp[i] : 0.0f);
      }
    }
    // Prefetch next K-tile while this one computes (global_prefetch_b8;
    // speculative, so harmless at the K tail / narrow-N edges).
    if (k0 + 32 < K) {
      __builtin_prefetch(A + (br + ar) * lda + (k0 + 32 + ach), 0, 1);
      __builtin_prefetch(B + (long)(k0 + 32 + bk) * ldb + bc + bnb, 0, 1);
    }
    __syncthreads();
    FragAB a0, a1, b0, b1;
#pragma unroll
    for (int v = 0; v < 8; ++v) {
      const int kk = kmap_(v, half);
      a0.u[v] = *(const unsigned*)&sA[wm + lrow][kk];
      a1.u[v] = *(const unsigned*)&sA[wm + 16 + lrow][kk];
      b0.u[v] = *(const unsigned*)&sB[wn + lrow][kk];
      b1.u[v] = *(const unsigned*)&sB[wn + 16 + lrow][kk];
    }
    acc[0][0] = __builtin_amdgcn_wmma_f32_16x16x32_bf16(false, a0.v, false, b0.v, (short)0, acc[0][0], false, false);
    acc[0][1] = __builtin_amdgcn_wmma_f32_16x16x32_bf16(false, a0.v, false, b1.v, (short)0, acc[0][1], false, false);
    acc[1][0] = __builtin_amdgcn_wmma_f32_16x16x32_bf16(false, a1.v, false, b0.v, (short)0, acc[1][0], false, false);
    acc[1][1] = __builtin_amdgcn_wmma_f32_16x16x32_bf16(false, a1.v, false, b1.v, (short)0, acc[1][1], false, false);
    __syncthreads();
  }

  // Epilogue. C/D layout: VGPR j -> row j + 8*half, col = lane&15 (per tile).
#pragma unroll
  for (int sm = 0; sm < 2; ++sm)
#pragma unroll
    for (int sn = 0; sn < 2; ++sn)
#pragma unroll
      for (int j = 0; j < 8; ++j) {
        const long row = br + wm + sm * 16 + j + 8 * half;
        const long col = bc + wn + sn * 16 + lrow;
        if (row < M && col < N) {
          float v = acc[sm][sn][j];
          if (EPI == 1 || EPI == 2 || EPI == 3) v += bias[col];
          const long idx = row * ldc + col;
          if (EPI == 0 || EPI == 1) C[idx] = v;
          else if (EPI == 2) C[idx] += v;
          else if (EPI == 3) C[idx] += aux[row * ldaux + col] * (1.0f / (1.0f + __expf(-v)));
        }
      }
}

// ---------------------------------------------------------------------------
// Fused attention: one workgroup per (b,h). 128 threads = 4 waves; wave w owns
// query rows [16w,16w+16). Scores stay in WMMA accumulators over the K=256
// loop; register softmax via shfl_xor within 16-lane halves; P@V via WMMA.
// q,k read from TMP [8192][2048] (post-rotation), v from QKV [8192][3072].
// mask_thresh: 6 for local (d<2.5 <=> d^2<=6), INT_MAX for global layers ->
// branch-free per-element v_cmp/v_cndmask (no uniform-branch ladder).
// ---------------------------------------------------------------------------
__global__ __launch_bounds__(128) void attn_wmma(
    const float* __restrict__ TMP, const float* __restrict__ QKV,
    float* __restrict__ OUT, int mask_thresh) {
  __shared__ __align__(8) unsigned short lq[64][36];
  __shared__ __align__(8) unsigned short lk[64][36];
  __shared__ __align__(8) unsigned short lp[64][68];   // probs bf16 [s][s']
  __shared__ __align__(8) unsigned short lvt[64][68];  // v chunk [d][s']
  const int b = blockIdx.x >> 2, h = blockIdx.x & 3;
  const int tid = threadIdx.x, lane = tid & 31, wave = tid >> 5;
  const int half = lane >> 4, lrow = lane & 15;
  const float* qp = TMP + (long)(b * 64) * 2048 + h * 256;
  const float* kp = qp + 1024;
  const float* vp = QKV + (long)(b * 64) * 3072 + 2048 + h * 256;
  float* op = OUT + (long)(b * 64) * 1024 + h * 256;

  f32x8 acc[4];
#pragma unroll
  for (int t = 0; t < 4; ++t) acc[t] = zero8();

  for (int kc = 0; kc < 8; ++kc) {  // hd=256 in chunks of 32
    const int r = tid >> 1, ch = (tid & 1) * 16;
    const float* q2 = qp + (long)r * 2048 + kc * 32 + ch;
    const float* k2 = kp + (long)r * 2048 + kc * 32 + ch;
#pragma unroll
    for (int i = 0; i < 16; ++i) { lq[r][ch + i] = f2bf(q2[i]); lk[r][ch + i] = f2bf(k2[i]); }
    if (kc < 7) {  // prefetch next hd chunk
      __builtin_prefetch(q2 + 32, 0, 1);
      __builtin_prefetch(k2 + 32, 0, 1);
    }
    __syncthreads();
    FragAB qa;
#pragma unroll
    for (int v = 0; v < 8; ++v) qa.u[v] = *(const unsigned*)&lq[wave * 16 + lrow][kmap_(v, half)];
#pragma unroll
    for (int t = 0; t < 4; ++t) {
      FragAB kb;
#pragma unroll
      for (int v = 0; v < 8; ++v) kb.u[v] = *(const unsigned*)&lk[t * 16 + lrow][kmap_(v, half)];
      acc[t] = __builtin_amdgcn_wmma_f32_16x16x32_bf16(false, qa.v, false, kb.v, (short)0, acc[t], false, false);
    }
    __syncthreads();
  }

  // mask + softmax (8 rows per lane: row = j + 8*half within wave's 16 rows)
  float rmax[8], rsum[8];
#pragma unroll
  for (int j = 0; j < 8; ++j) { rmax[j] = -3e38f; rsum[j] = 0.f; }
#pragma unroll
  for (int t = 0; t < 4; ++t)
#pragma unroll
    for (int j = 0; j < 8; ++j) {
      const int sr = wave * 16 + j + 8 * half, sc = t * 16 + lrow;
      const int dy = (sr >> 3) - (sc >> 3), dx = (sr & 7) - (sc & 7);
      const int d2 = dy * dy + dx * dx;
      float v = acc[t][j] * 0.0625f;                 // HD^-0.5
      v = (d2 <= mask_thresh) ? v : -1e30f;          // branch-free mask
      acc[t][j] = v;
      rmax[j] = fmaxf(rmax[j], v);
    }
#pragma unroll
  for (int j = 0; j < 8; ++j)
    for (int m = 1; m < 16; m <<= 1) rmax[j] = fmaxf(rmax[j], __shfl_xor(rmax[j], m, 32));
#pragma unroll
  for (int t = 0; t < 4; ++t)
#pragma unroll
    for (int j = 0; j < 8; ++j) { float e = __expf(acc[t][j] - rmax[j]); acc[t][j] = e; rsum[j] += e; }
#pragma unroll
  for (int j = 0; j < 8; ++j)
    for (int m = 1; m < 16; m <<= 1) rsum[j] += __shfl_xor(rsum[j], m, 32);
#pragma unroll
  for (int t = 0; t < 4; ++t)
#pragma unroll
    for (int j = 0; j < 8; ++j)
      lp[wave * 16 + j + 8 * half][t * 16 + lrow] = f2bf(acc[t][j] / rsum[j]);
  __syncthreads();

  // P[64x64] @ V[64x256] in 4 column chunks of 64
  for (int nc = 0; nc < 4; ++nc) {
    const int sr = tid >> 1, dh = (tid & 1) * 32;
    const float* v2 = vp + (long)sr * 3072 + nc * 64 + dh;
#pragma unroll
    for (int i = 0; i < 32; ++i) lvt[dh + i][sr] = f2bf(v2[i]);  // transpose to [d][s']
    if (nc < 3) __builtin_prefetch(v2 + 64, 0, 1);
    __syncthreads();
    f32x8 o[4];
#pragma unroll
    for (int t = 0; t < 4; ++t) o[t] = zero8();
#pragma unroll
    for (int ks = 0; ks < 64; ks += 32) {
      FragAB pa;
#pragma unroll
      for (int v = 0; v < 8; ++v) pa.u[v] = *(const unsigned*)&lp[wave * 16 + lrow][ks + kmap_(v, half)];
#pragma unroll
      for (int t = 0; t < 4; ++t) {
        FragAB vb;
#pragma unroll
        for (int v = 0; v < 8; ++v) vb.u[v] = *(const unsigned*)&lvt[t * 16 + lrow][ks + kmap_(v, half)];
        o[t] = __builtin_amdgcn_wmma_f32_16x16x32_bf16(false, pa.v, false, vb.v, (short)0, o[t], false, false);
      }
    }
#pragma unroll
    for (int t = 0; t < 4; ++t)
#pragma unroll
      for (int j = 0; j < 8; ++j)
        op[(long)(wave * 16 + j + 8 * half) * 1024 + nc * 64 + t * 16 + lrow] = o[t][j];
    __syncthreads();
  }
}

// ---------------------------------------------------------------------------
// Small kernels
// ---------------------------------------------------------------------------
struct Ptr8 { const float* p[8]; };

// Sequential Householder product per layer; thread j owns column j of Q
// (w_j = v.Q[:,j] and the rank-1 update touch only column j -> no Q hazards).
__global__ __launch_bounds__(256) void house_q(Ptr8 vs, float* Qb, float* QTb) {
  const int L = blockIdx.x, j = threadIdx.x;
  float* Q  = Qb  + (long)L * 65536;
  float* QT = QTb + (long)L * 65536;
  for (int i = 0; i < 256; ++i) Q[i * 256 + j] = (i == j) ? 1.f : 0.f;
  __shared__ float v[256];
  const float* rv = vs.p[L];
  for (int s = 0; s < 128; ++s) {
    __syncthreads();
    v[j] = rv[s * 256 + j];
    __syncthreads();
    float vv = 0.f, wj = 0.f;
    for (int i = 0; i < 256; ++i) { vv += v[i] * v[i]; wj += v[i] * Q[i * 256 + j]; }
    const float c = 2.f / (vv + 1e-8f);
    for (int i = 0; i < 256; ++i) Q[i * 256 + j] -= c * v[i] * wj;
  }
  __syncthreads();
  for (int i = 0; i < 256; ++i) QT[j * 256 + i] = Q[i * 256 + j];
}

__global__ __launch_bounds__(256) void rms_rows(const float* __restrict__ X, float* __restrict__ Y) {
  const long r = blockIdx.x; const int t = threadIdx.x;
  __shared__ float red[256];
  const float* x = X + r * 1024;
  const float v0 = x[t], v1 = x[t + 256], v2 = x[t + 512], v3 = x[t + 768];
  red[t] = v0 * v0 + v1 * v1 + v2 * v2 + v3 * v3;
  __syncthreads();
  for (int s = 128; s > 0; s >>= 1) { if (t < s) red[t] += red[t + s]; __syncthreads(); }
  const float inv = rsqrtf(red[0] * (1.f / 1024.f) + 1.1920929e-7f);
  float* y = Y + r * 1024;
  y[t] = v0 * inv; y[t + 256] = v1 * inv; y[t + 512] = v2 * inv; y[t + 768] = v3 * inv;
}

__global__ void fourier_spins(const float* __restrict__ logsnr, float* __restrict__ spins) {
  const int b = threadIdx.x;
  if (b < 128) {
    const float x = logsnr[b];
#pragma unroll
    for (int i = 0; i < 4; ++i) {
      const float a = x * (float)(1 << i);
      spins[b * 8 + i] = sinf(a);
      spins[b * 8 + 4 + i] = cosf(a);
    }
  }
}

__global__ __launch_bounds__(256) void rope_tables(float* __restrict__ COS, float* __restrict__ SIN) {
  const int s = blockIdx.x, d = threadIdx.x;
  const int u = d & 127, t = u & 63;
  const float coord = (u < 64) ? (float)(s >> 3) : (float)(s & 7);
  const float inv = powf(10000.f, -(float)t / 64.f);
  const float a = coord * inv;
  COS[s * 256 + d] = cosf(a);
  SIN[s * 256 + d] = sinf(a);
}

__global__ __launch_bounds__(256) void patch_embed(
    const float* __restrict__ z, const float* __restrict__ spins,
    const float* __restrict__ W, const float* __restrict__ bias, float* __restrict__ X) {
  const int r = blockIdx.y;
  const int c = blockIdx.x * 256 + threadIdx.x;
  const int b = r >> 6, s = r & 63, yy = s >> 3, xx = s & 7;
  __shared__ float sv[20];
  if (threadIdx.x < 20) {
    const int k = threadIdx.x;
    if (k < 12) {
      const int cch = k >> 2, py = (k >> 1) & 1, px = k & 1;
      sv[k] = z[((b * 3 + cch) * 16 + (2 * yy + py)) * 16 + (2 * xx + px)];
    } else sv[k] = spins[b * 8 + (k - 12)];
  }
  __syncthreads();
  float a = bias[c];
#pragma unroll
  for (int k = 0; k < 20; ++k) a += sv[k] * W[k * 1024 + c];
  X[(long)r * 1024 + c] = a;
}

// RoPE rotate: TMP[8192][2048] (q|k after @Q^T) -> QKV q,k slots (same col idx)
__global__ __launch_bounds__(256) void rope_rotate(
    const float* __restrict__ T, const float* __restrict__ COS,
    const float* __restrict__ SIN, float* __restrict__ QKV) {
  const long idx = (long)blockIdx.x * 256 + threadIdx.x;
  const long r = idx >> 11; const int c = (int)(idx & 2047);
  const int d = c & 255, s = (int)(r & 63);
  const float t1 = T[idx];
  const int c2 = (d < 128) ? (c + 128) : (c - 128);
  const float t2 = T[(r << 11) + c2];
  QKV[r * 3072 + c] = t1 * COS[s * 256 + d] + ((d < 128) ? -t2 : t2) * SIN[s * 256 + d];
}

// act = val * gelu(g) (exact), writes over g (R1)
__global__ __launch_bounds__(256) void gelu_val(float* __restrict__ G, const float* __restrict__ V) {
  const long idx = (long)blockIdx.x * 256 + threadIdx.x;
  const float g = G[idx];
  G[idx] = V[idx] * 0.5f * g * (1.0f + erff(g * 0.70710678118654752f));
}

__global__ __launch_bounds__(128) void scale_head(
    const float* __restrict__ spins, const float* __restrict__ w1, const float* __restrict__ b1,
    const float* __restrict__ w2, const float* __restrict__ b2, float* __restrict__ scale) {
  const int b = blockIdx.x;
  __shared__ float s1[1024];
  for (int c = threadIdx.x; c < 1024; c += 128) {
    float t = b1[c];
#pragma unroll
    for (int k = 0; k < 8; ++k) t += spins[b * 8 + k] * w1[k * 1024 + c];
    s1[c] = t / (1.0f + __expf(-t));  // silu
  }
  __syncthreads();
  if (threadIdx.x < 12) {
    float a = b2[threadIdx.x];
    for (int c = 0; c < 1024; ++c) a += s1[c] * w2[c * 12 + threadIdx.x];
    scale[b * 12 + threadIdx.x] = __expf(a);
  }
}

__global__ __launch_bounds__(256) void unpatch(
    const float* __restrict__ zp, const float* __restrict__ scale, float* __restrict__ out) {
  const int idx = blockIdx.x * 256 + threadIdx.x;  // B*3*16*16 = 98304
  const int j = idx & 15, i = (idx >> 4) & 15, ch = (idx >> 8) % 3, b = idx / 768;
  const int s = (i >> 1) * 8 + (j >> 1);
  const int k = ch * 4 + (i & 1) * 2 + (j & 1);
  out[idx] = zp[(b * 64 + s) * 12 + k] * scale[b * 12 + k];
}

__global__ __launch_bounds__(256) void lam_head(
    const float* __restrict__ H, const float* __restrict__ lw, const float* __restrict__ lb,
    float* __restrict__ out) {
  const int b = blockIdx.x, t = threadIdx.x;
  __shared__ float red[256];
  float p = 0.f;
  for (int c = t; c < 1024; c += 256) {
    float m = 0.f;
    for (int s = 0; s < 64; ++s) m += H[(long)(b * 64 + s) * 1024 + c];
    p += (m * (1.f / 64.f)) * lw[c];
  }
  red[t] = p;
  __syncthreads();
  for (int s = 128; s > 0; s >>= 1) { if (t < s) red[t] += red[t + s]; __syncthreads(); }
  if (t == 0) out[98304 + b] = red[0] + lb[0];
}

// ---------------------------------------------------------------------------
// Host orchestration
// ---------------------------------------------------------------------------
static void launch_gemm(int epi, const float* A, long lda, const float* B, long ldb,
                        const float* bias, float* C, long ldc, const float* aux, long ldaux,
                        int M, int N, int K, hipStream_t s) {
  dim3 g((unsigned)((N + 63) / 64), (unsigned)((M + 63) / 64)), b(128);
  switch (epi) {
    case 0: gemm_bf16_wmma<0><<<g, b, 0, s>>>(A, lda, B, ldb, bias, C, ldc, aux, ldaux, M, N, K); break;
    case 1: gemm_bf16_wmma<1><<<g, b, 0, s>>>(A, lda, B, ldb, bias, C, ldc, aux, ldaux, M, N, K); break;
    case 2: gemm_bf16_wmma<2><<<g, b, 0, s>>>(A, lda, B, ldb, bias, C, ldc, aux, ldaux, M, N, K); break;
    default: gemm_bf16_wmma<3><<<g, b, 0, s>>>(A, lda, B, ldb, bias, C, ldc, aux, ldaux, M, N, K); break;
  }
}

extern "C" void kernel_launch(void* const* d_in, const int* in_sizes, int n_in,
                              void* d_out, int out_size, void* d_ws, size_t ws_size,
                              hipStream_t stream) {
  if (n_in < 84) return;
  const float* in[84];
  for (int i = 0; i < 84; ++i) in[i] = (const float*)d_in[i];

  const float *z_t, *logsnr, *piw, *pib, *ow, *ob, *sw1, *sb1, *sw2, *sb2, *lamw, *lamb;
  const float *Lqkv[8], *Lout[8], *Lgw[8], *Lgb[8], *Lmgw[8], *Lmgb[8], *Lmow[8], *Lmob[8], *Lrope[8];

  if (in_sizes[0] == 98304) {  // insertion order: z_t, logsnr, params{...}
    z_t = in[0]; logsnr = in[1]; piw = in[2]; pib = in[3];
    for (int l = 0; l < 8; ++l) {
      const int o = 4 + l * 9;
      Lqkv[l] = in[o]; Lout[l] = in[o + 1]; Lgw[l] = in[o + 2]; Lgb[l] = in[o + 3];
      Lmgw[l] = in[o + 4]; Lmgb[l] = in[o + 5]; Lmow[l] = in[o + 6]; Lmob[l] = in[o + 7];
      Lrope[l] = in[o + 8];
    }
    ow = in[76]; ob = in[77]; sw1 = in[78]; sb1 = in[79]; sw2 = in[80]; sb2 = in[81];
    lamw = in[82]; lamb = in[83];
  } else {  // jax pytree alphabetical order
    logsnr = in[0]; lamb = in[1]; lamw = in[2];
    for (int l = 0; l < 8; ++l) {
      const int o = 3 + l * 9;
      Lgb[l] = in[o]; Lgw[l] = in[o + 1]; Lmgb[l] = in[o + 2]; Lmgw[l] = in[o + 3];
      Lmob[l] = in[o + 4]; Lmow[l] = in[o + 5]; Lout[l] = in[o + 6]; Lqkv[l] = in[o + 7];
      Lrope[l] = in[o + 8];
    }
    ob = in[75]; ow = in[76]; pib = in[77]; piw = in[78];
    sb1 = in[79]; sb2 = in[80]; sw1 = in[81]; sw2 = in[82]; z_t = in[83];
  }

  // workspace layout (floats)
  float* W = (float*)d_ws;
  size_t o = 0;
  float* SPINS = W + o; o += 1024;
  float* COS   = W + o; o += 16384;
  float* SIN   = W + o; o += 16384;
  float* Q8    = W + o; o += 8u * 65536;
  float* QT8   = W + o; o += 8u * 65536;
  float* X     = W + o; o += 8388608;           // [8192][1024]
  float* H     = W + o; o += 8388608;           // rms output
  float* QKV   = W + o; o += 25165824;          // [8192][3072]
  float* R1    = W + o; o += 33554432;          // MLP g / attention temps
  float* R2    = W + o; o += 33554432;          // MLP val
  float* ZPRED = W + o; o += 98304;             // [8192][12]
  float* SCALE = W + o; o += 1536;              // [128][12]
  float* TMP   = R1;                            // [8192][2048] rotation temp
  float* ATTN  = R1 + 16777216;                 // [8192][1024]
  float* APROJ = R1 + 25165824;                 // [8192][1024]
  (void)ws_size; (void)out_size;

  // --- setup ---
  fourier_spins<<<1, 128, 0, stream>>>(logsnr, SPINS);
  rope_tables<<<64, 256, 0, stream>>>(COS, SIN);
  Ptr8 rp; for (int l = 0; l < 8; ++l) rp.p[l] = Lrope[l];
  house_q<<<8, 256, 0, stream>>>(rp, Q8, QT8);
  patch_embed<<<dim3(4, 8192), 256, 0, stream>>>(z_t, SPINS, piw, pib, X);

  // --- transformer layers ---
  for (int l = 0; l < 8; ++l) {
    const float* Ql = Q8 + (size_t)l * 65536;
    const float* QTl = QT8 + (size_t)l * 65536;

    rms_rows<<<8192, 256, 0, stream>>>(X, H);
    launch_gemm(0, H, 1024, Lqkv[l], 3072, nullptr, QKV, 3072, nullptr, 0,
                8192, 3072, 1024, stream);

    // q,k @ Q^T (per head, strided views)
    for (int h = 0; h < 4; ++h) {
      launch_gemm(0, QKV + h * 256, 3072, QTl, 256, nullptr, TMP + h * 256, 2048,
                  nullptr, 0, 8192, 256, 256, stream);
      launch_gemm(0, QKV + 1024 + h * 256, 3072, QTl, 256, nullptr,
                  TMP + 1024 + h * 256, 2048, nullptr, 0, 8192, 256, 256, stream);
    }
    rope_rotate<<<65536, 256, 0, stream>>>(TMP, COS, SIN, QKV);  // -> QKV q,k slots
    // rot(q), rot(k) @ Q
    for (int h = 0; h < 4; ++h) {
      launch_gemm(0, QKV + h * 256, 3072, Ql, 256, nullptr, TMP + h * 256, 2048,
                  nullptr, 0, 8192, 256, 256, stream);
      launch_gemm(0, QKV + 1024 + h * 256, 3072, Ql, 256, nullptr,
                  TMP + 1024 + h * 256, 2048, nullptr, 0, 8192, 256, 256, stream);
    }

    const int thresh = (((l + 1) % 4) == 0) ? 0x7FFFFFFF : 6;
    attn_wmma<<<512, 128, 0, stream>>>(TMP, QKV, ATTN, thresh);

    launch_gemm(0, ATTN, 1024, Lout[l], 1024, nullptr, APROJ, 1024, nullptr, 0,
                8192, 1024, 1024, stream);
    // x += aproj * sigmoid(aproj @ gate_w + gate_b)   (fused epilogue)
    launch_gemm(3, APROJ, 1024, Lgw[l], 1024, Lgb[l], X, 1024, APROJ, 1024,
                8192, 1024, 1024, stream);

    rms_rows<<<8192, 256, 0, stream>>>(X, H);
    // g = cols [0,4096), val = cols [4096,8192) of mlpg
    launch_gemm(1, H, 1024, Lmgw[l], 8192, Lmgb[l], R1, 4096, nullptr, 0,
                8192, 4096, 1024, stream);
    launch_gemm(1, H, 1024, Lmgw[l] + 4096, 8192, Lmgb[l] + 4096, R2, 4096,
                nullptr, 0, 8192, 4096, 1024, stream);
    gelu_val<<<131072, 256, 0, stream>>>(R1, R2);     // R1 = val * gelu(g)
    // x += act @ mlpo_w + mlpo_b   (residual epilogue)
    launch_gemm(2, R1, 4096, Lmow[l], 1024, Lmob[l], X, 1024, nullptr, 0,
                8192, 1024, 4096, stream);
  }

  // --- heads ---
  rms_rows<<<8192, 256, 0, stream>>>(X, H);
  launch_gemm(1, H, 1024, ow, 12, ob, ZPRED, 12, nullptr, 0, 8192, 12, 1024, stream);
  scale_head<<<128, 128, 0, stream>>>(SPINS, sw1, sb1, sw2, sb2, SCALE);
  unpatch<<<384, 256, 0, stream>>>(ZPRED, SCALE, (float*)d_out);
  lam_head<<<128, 256, 0, stream>>>(H, lamw, lamb, (float*)d_out);
}
